// LIFModule_17454747091638
// MI455X (gfx1250) — matrix operand order repaired and checked by reading, not compiled, not run on hardware
//
#include <hip/hip_runtime.h>

typedef __bf16 bf16;
typedef __attribute__((ext_vector_type(16))) __bf16 v16bf;
typedef __attribute__((ext_vector_type(8)))  float  v8f;

#define C96   96
#define HW    3136            // 56*56
#define PER_B (C96*HW)        // 301056 elements per sample
#define NTOT  (32*PER_B)      // 9633792 elements per tensor
#define PXT   64              // pixels per block (3136 % 64 == 0)
#define GEPS  1e-5f

__device__ __forceinline__ float gelu_exact(float x) {
    return 0.5f * x * (1.0f + erff(x * 0.70710678118654752f));
}

// ---------------------------------------------------------------- utilities

__global__ __launch_bounds__(256) void k_init(float* stats) {
    stats[threadIdx.x] = 0.0f;   // 256 floats: 3 GN stages at +0, +64, +128
}

// Pack f32 96x96 weights -> bf16 in WMMA A-fragment order:
//   dst[m][ ((otile*3 + kc)*32 + (M + 16*halfA))*16 + jA ]
//   halfA = (c>>3)&1, jA = (c&7) + 8*((c>>4)&1)   (16-bit A 16x32 ISA layout)
__global__ __launch_bounds__(256) void k_pack_w(const float* __restrict__ w1,
                                                const float* __restrict__ w21,
                                                const float* __restrict__ w22,
                                                const float* __restrict__ w3,
                                                bf16* __restrict__ dst) {
    int i = blockIdx.x * 256 + threadIdx.x;
    if (i >= 4 * 9216) return;
    const float* src = (i < 9216) ? w1 : (i < 18432) ? w21 : (i < 27648) ? w22 : w3;
    int m = i / 9216;
    int r = i % 9216;
    int o = r / C96, c = r % C96;
    int off = (((o >> 4) * 3 + (c >> 5)) * 32 + ((o & 15) + 16 * ((c >> 3) & 1))) * 16
              + ((c & 7) + 8 * ((c >> 4) & 1));
    dst[m * 9216 + off] = (bf16)src[r];
}

// Two-stage GN stats: per-(sample, slice) partial sums -> atomicAdd
__global__ __launch_bounds__(256) void k_reduce(const float* __restrict__ X,
                                                float* __restrict__ stats) {
    int b = blockIdx.x;
    size_t base = (size_t)b * PER_B + (size_t)blockIdx.y * 6144;  // 49 slices * 6144
    float s = 0.0f, ss = 0.0f;
    for (int i = threadIdx.x; i < 6144; i += 256) {
        float v = X[base + i];
        s += v; ss += v * v;
    }
    __shared__ float rs[256], rq[256];
    rs[threadIdx.x] = s; rq[threadIdx.x] = ss;
    __syncthreads();
    for (int o = 128; o > 0; o >>= 1) {
        if (threadIdx.x < o) {
            rs[threadIdx.x] += rs[threadIdx.x + o];
            rq[threadIdx.x] += rq[threadIdx.x + o];
        }
        __syncthreads();
    }
    if (threadIdx.x == 0) {
        atomicAdd(&stats[b * 2 + 0], rs[0]);
        atomicAdd(&stats[b * 2 + 1], rq[0]);
    }
}

__global__ __launch_bounds__(256) void k_gngelu(const float* __restrict__ X,
                                                const float* __restrict__ stats,
                                                const float* __restrict__ gw,
                                                const float* __restrict__ gb,
                                                float* Y) {
    int i = blockIdx.x * 256 + threadIdx.x;
    if (i >= NTOT) return;
    int b = i / PER_B;
    int c = (i / HW) % C96;
    float mu   = stats[b * 2 + 0] * (1.0f / PER_B);
    float var  = stats[b * 2 + 1] * (1.0f / PER_B) - mu * mu;
    float rsig = rsqrtf(var + GEPS);
    float v = (X[i] - mu) * rsig * gw[c] + gb[c];
    Y[i] = gelu_exact(v);
}

__global__ __launch_bounds__(256) void k_dw3x3(const float* __restrict__ X,
                                               const float* __restrict__ W,
                                               const float* __restrict__ bias,
                                               float* __restrict__ Y) {
    int i = blockIdx.x * 256 + threadIdx.x;
    if (i >= NTOT) return;
    int w = i % 56;
    int h = (i / 56) % 56;
    int c = (i / HW) % C96;
    int b = i / PER_B;
    const float* xp = X + (size_t)(b * C96 + c) * HW;
    const float* wp = W + c * 9;
    float acc = bias[c];
    #pragma unroll
    for (int ky = 0; ky < 3; ++ky) {
        int hh = h + ky - 1;
        if ((unsigned)hh < 56u) {
            #pragma unroll
            for (int kx = 0; kx < 3; ++kx) {
                int ww = w + kx - 1;
                if ((unsigned)ww < 56u)
                    acc += wp[ky * 3 + kx] * xp[hh * 56 + ww];
            }
        }
    }
    Y[i] = acc;
}

// LIF scan, 4-step chunks; axis 0 = along H (stride 56), axis 1 = along W (stride 1)
__global__ __launch_bounds__(256) void k_lif(const float* __restrict__ X,
                                             const float* __restrict__ tau_,
                                             const float* __restrict__ vth_,
                                             float* __restrict__ Y, int axis) {
    const int n = 32 * C96 * 14 * 56;
    int i = blockIdx.x * 256 + threadIdx.x;
    if (i >= n) return;
    float tau = tau_[0], vth = vth_[0];
    size_t base; int stride;
    if (axis == 0) {
        int w  = i % 56;
        int g  = (i / 56) % 14;
        int cb = i / (56 * 14);                 // b*96 + c
        base = (size_t)cb * HW + (g * 4) * 56 + w;
        stride = 56;
    } else {
        int g  = i % 14;
        int h  = (i / 14) % 56;
        int cb = i / (14 * 56);
        base = (size_t)cb * HW + h * 56 + g * 4;
        stride = 1;
    }
    float u = 0.0f, s = 0.0f;
    #pragma unroll
    for (int t = 0; t < 4; ++t) {
        float x = X[base + (size_t)t * stride];
        u = x + tau * u * (1.0f - s);
        float sp = (u > vth) ? 1.0f : 0.0f;
        Y[base + (size_t)t * stride] = u * sp;
        s = sp;
    }
}

// --------------------------------------------------------- WMMA GEMM helpers
// LDS tile layout = B-fragment order: frag[sub][kc][lane][j], j contiguous.
//   B 32x16 layout: lane = px + 16*((klocal>>3)&1), j = (klocal&7) + 8*(klocal>>4).
// Writer packs channel-pairs (c even, c+1) into one 32-bit LDS store.

#define WMMA_BF16(A, B, C) \
    __builtin_amdgcn_wmma_f32_16x16x32_bf16(false, (A), false, (B), (short)0, (C), false, false)

__device__ __forceinline__ int frag_idx(int c, int pxg) {
    // returns element index (even j) for channel pair (c, c+1), pixel pxg in [0,64)
    return (((pxg >> 4) * 3 + (c >> 5)) * 32 + (pxg & 15) + 16 * ((c >> 3) & 1)) * 16
           + ((c & 7) + 8 * ((c >> 4) & 1));
}

__global__ __launch_bounds__(192) void k_conv1(const float* __restrict__ X,
                                               const bf16* __restrict__ W,
                                               const float* __restrict__ bias,
                                               float* __restrict__ Y) {
    __shared__ __align__(32) bf16 frag[4 * 3 * 32 * 16];   // 12 KB
    const int tid  = threadIdx.x;
    const int wave = tid >> 5, lane = tid & 31;
    const int g0  = blockIdx.x * PXT;
    const int b   = g0 / HW;
    const int hw0 = g0 % HW;
    __builtin_prefetch(W + ((wave * 3) * 32 + lane) * 16, 0, 0);
    const float* xb = X + (size_t)(b * C96) * HW + hw0;
    for (int i = tid; i < 48 * PXT; i += 192) {            // 3072 channel-pair stores
        int pxg = i & (PXT - 1);
        int c   = (i >> 6) * 2;
        union { bf16 h[2]; unsigned u; } p;
        p.h[0] = (bf16)xb[c * HW + pxg];
        p.h[1] = (bf16)xb[(c + 1) * HW + pxg];
        *(unsigned*)&frag[frag_idx(c, pxg)] = p.u;
    }
    __syncthreads();
    const v16bf a0 = *(const v16bf*)(W + ((wave * 3 + 0) * 32 + lane) * 16);
    const v16bf a1 = *(const v16bf*)(W + ((wave * 3 + 1) * 32 + lane) * 16);
    const v16bf a2 = *(const v16bf*)(W + ((wave * 3 + 2) * 32 + lane) * 16);
    const int mrow = lane & 15;
    const int half = lane >> 4;
    float bv[8];
    #pragma unroll
    for (int r = 0; r < 8; ++r) bv[r] = bias[wave * 16 + r + half * 8];
    #pragma unroll
    for (int sub = 0; sub < 4; ++sub) {
        const bf16* fb = frag + sub * (3 * 32 * 16);
        v8f acc;
        #pragma unroll
        for (int r = 0; r < 8; ++r) acc[r] = bv[r];
        acc = WMMA_BF16(a0, *(const v16bf*)(fb + (0 * 32 + lane) * 16), acc);
        acc = WMMA_BF16(a1, *(const v16bf*)(fb + (1 * 32 + lane) * 16), acc);
        acc = WMMA_BF16(a2, *(const v16bf*)(fb + (2 * 32 + lane) * 16), acc);
        float* yb = Y + (size_t)(b * C96 + wave * 16) * HW + hw0 + sub * 16;
        #pragma unroll
        for (int r = 0; r < 8; ++r)
            yb[(size_t)(r + half * 8) * HW + mrow] = acc[r];
    }
}

// Dual GEMM: z = gelu(W21 @ x_lr + b21) + gelu(W22 @ x_td + b22)
__global__ __launch_bounds__(192) void k_conv2(const float* __restrict__ Xlr,
                                               const float* __restrict__ Xtd,
                                               const bf16* __restrict__ W21,
                                               const bf16* __restrict__ W22,
                                               const float* __restrict__ b21,
                                               const float* __restrict__ b22,
                                               float* __restrict__ Z) {
    __shared__ __align__(32) bf16 f1[4 * 3 * 32 * 16];     // 12 KB
    __shared__ __align__(32) bf16 f2[4 * 3 * 32 * 16];     // 12 KB
    const int tid  = threadIdx.x;
    const int wave = tid >> 5, lane = tid & 31;
    const int g0  = blockIdx.x * PXT;
    const int b   = g0 / HW;
    const int hw0 = g0 % HW;
    __builtin_prefetch(W21 + ((wave * 3) * 32 + lane) * 16, 0, 0);
    __builtin_prefetch(W22 + ((wave * 3) * 32 + lane) * 16, 0, 0);
    const float* x1 = Xlr + (size_t)(b * C96) * HW + hw0;
    const float* x2 = Xtd + (size_t)(b * C96) * HW + hw0;
    for (int i = tid; i < 48 * PXT; i += 192) {
        int pxg = i & (PXT - 1);
        int c   = (i >> 6) * 2;
        int idx = frag_idx(c, pxg);
        union { bf16 h[2]; unsigned u; } p, q;
        p.h[0] = (bf16)x1[c * HW + pxg];
        p.h[1] = (bf16)x1[(c + 1) * HW + pxg];
        q.h[0] = (bf16)x2[c * HW + pxg];
        q.h[1] = (bf16)x2[(c + 1) * HW + pxg];
        *(unsigned*)&f1[idx] = p.u;
        *(unsigned*)&f2[idx] = q.u;
    }
    __syncthreads();
    const v16bf a10 = *(const v16bf*)(W21 + ((wave * 3 + 0) * 32 + lane) * 16);
    const v16bf a11 = *(const v16bf*)(W21 + ((wave * 3 + 1) * 32 + lane) * 16);
    const v16bf a12 = *(const v16bf*)(W21 + ((wave * 3 + 2) * 32 + lane) * 16);
    const v16bf a20 = *(const v16bf*)(W22 + ((wave * 3 + 0) * 32 + lane) * 16);
    const v16bf a21 = *(const v16bf*)(W22 + ((wave * 3 + 1) * 32 + lane) * 16);
    const v16bf a22 = *(const v16bf*)(W22 + ((wave * 3 + 2) * 32 + lane) * 16);
    const int mrow = lane & 15;
    const int half = lane >> 4;
    float bv1[8], bv2[8];
    #pragma unroll
    for (int r = 0; r < 8; ++r) {
        bv1[r] = b21[wave * 16 + r + half * 8];
        bv2[r] = b22[wave * 16 + r + half * 8];
    }
    #pragma unroll
    for (int sub = 0; sub < 4; ++sub) {
        const bf16* p1 = f1 + sub * (3 * 32 * 16);
        const bf16* p2 = f2 + sub * (3 * 32 * 16);
        v8f acc1, acc2;
        #pragma unroll
        for (int r = 0; r < 8; ++r) { acc1[r] = bv1[r]; acc2[r] = bv2[r]; }
        acc1 = WMMA_BF16(a10, *(const v16bf*)(p1 + (0 * 32 + lane) * 16), acc1);
        acc2 = WMMA_BF16(a20, *(const v16bf*)(p2 + (0 * 32 + lane) * 16), acc2);
        acc1 = WMMA_BF16(a11, *(const v16bf*)(p1 + (1 * 32 + lane) * 16), acc1);
        acc2 = WMMA_BF16(a21, *(const v16bf*)(p2 + (1 * 32 + lane) * 16), acc2);
        acc1 = WMMA_BF16(a12, *(const v16bf*)(p1 + (2 * 32 + lane) * 16), acc1);
        acc2 = WMMA_BF16(a22, *(const v16bf*)(p2 + (2 * 32 + lane) * 16), acc2);
        float* zb = Z + (size_t)(b * C96 + wave * 16) * HW + hw0 + sub * 16;
        #pragma unroll
        for (int r = 0; r < 8; ++r)
            zb[(size_t)(r + half * 8) * HW + mrow] =
                gelu_exact(acc1[r]) + gelu_exact(acc2[r]);
    }
}

// GN3 fused into tile load, then conv3 GEMM
__global__ __launch_bounds__(192) void k_conv3(const float* __restrict__ Zin,
                                               const float* __restrict__ stats,
                                               const float* __restrict__ nw,
                                               const float* __restrict__ nb,
                                               const bf16* __restrict__ W,
                                               const float* __restrict__ bias,
                                               float* __restrict__ Y) {
    __shared__ __align__(32) bf16 frag[4 * 3 * 32 * 16];
    const int tid  = threadIdx.x;
    const int wave = tid >> 5, lane = tid & 31;
    const int g0  = blockIdx.x * PXT;
    const int b   = g0 / HW;
    const int hw0 = g0 % HW;
    __builtin_prefetch(W + ((wave * 3) * 32 + lane) * 16, 0, 0);
    float mu   = stats[b * 2 + 0] * (1.0f / PER_B);
    float var  = stats[b * 2 + 1] * (1.0f / PER_B) - mu * mu;
    float rsig = rsqrtf(var + GEPS);
    const float* zb = Zin + (size_t)(b * C96) * HW + hw0;
    for (int i = tid; i < 48 * PXT; i += 192) {
        int pxg = i & (PXT - 1);
        int c   = (i >> 6) * 2;
        float v0 = (zb[c * HW + pxg] - mu) * rsig * nw[c] + nb[c];
        float v1 = (zb[(c + 1) * HW + pxg] - mu) * rsig * nw[c + 1] + nb[c + 1];
        union { bf16 h[2]; unsigned u; } p;
        p.h[0] = (bf16)v0; p.h[1] = (bf16)v1;
        *(unsigned*)&frag[frag_idx(c, pxg)] = p.u;
    }
    __syncthreads();
    const v16bf a0 = *(const v16bf*)(W + ((wave * 3 + 0) * 32 + lane) * 16);
    const v16bf a1 = *(const v16bf*)(W + ((wave * 3 + 1) * 32 + lane) * 16);
    const v16bf a2 = *(const v16bf*)(W + ((wave * 3 + 2) * 32 + lane) * 16);
    const int mrow = lane & 15;
    const int half = lane >> 4;
    float bv[8];
    #pragma unroll
    for (int r = 0; r < 8; ++r) bv[r] = bias[wave * 16 + r + half * 8];
    #pragma unroll
    for (int sub = 0; sub < 4; ++sub) {
        const bf16* fb = frag + sub * (3 * 32 * 16);
        v8f acc;
        #pragma unroll
        for (int r = 0; r < 8; ++r) acc[r] = bv[r];
        acc = WMMA_BF16(a0, *(const v16bf*)(fb + (0 * 32 + lane) * 16), acc);
        acc = WMMA_BF16(a1, *(const v16bf*)(fb + (1 * 32 + lane) * 16), acc);
        acc = WMMA_BF16(a2, *(const v16bf*)(fb + (2 * 32 + lane) * 16), acc);
        float* yb = Y + (size_t)(b * C96 + wave * 16) * HW + hw0 + sub * 16;
        #pragma unroll
        for (int r = 0; r < 8; ++r)
            yb[(size_t)(r + half * 8) * HW + mrow] = acc[r];
    }
}

// ------------------------------------------------------------------- launch

extern "C" void kernel_launch(void* const* d_in, const int* in_sizes, int n_in,
                              void* d_out, int out_size, void* d_ws, size_t ws_size,
                              hipStream_t stream) {
    const float* x    = (const float*)d_in[0];
    const float* w1   = (const float*)d_in[1];
    const float* b1   = (const float*)d_in[2];
    const float* n1w  = (const float*)d_in[3];
    const float* n1b  = (const float*)d_in[4];
    const float* dww  = (const float*)d_in[5];
    const float* dwb  = (const float*)d_in[6];
    const float* n2w  = (const float*)d_in[7];
    const float* n2b  = (const float*)d_in[8];
    const float* tau1 = (const float*)d_in[9];
    const float* vth1 = (const float*)d_in[10];
    const float* tau2 = (const float*)d_in[11];
    const float* vth2 = (const float*)d_in[12];
    const float* w21  = (const float*)d_in[13];
    const float* b21  = (const float*)d_in[14];
    const float* w22  = (const float*)d_in[15];
    const float* b22  = (const float*)d_in[16];
    const float* n3w  = (const float*)d_in[17];
    const float* n3b  = (const float*)d_in[18];
    const float* w3   = (const float*)d_in[19];
    const float* b3   = (const float*)d_in[20];
    float* out = (float*)d_out;

    char*  ws    = (char*)d_ws;
    float* stats = (float*)ws;                      // 256 floats
    bf16*  wb    = (bf16*)(ws + 4096);              // 4 * 96*96 bf16, A-fragment order
    float* A     = (float*)(ws + (1 << 17));        // 3 tensor-sized buffers
    float* B     = A + NTOT;
    float* Cc    = B + NTOT;

    const int ntiles = (32 * HW) / PXT;             // 1568 blocks of 64 pixels
    const int eb     = (NTOT + 255) / 256;
    const dim3 rg(32, 49);                          // 49 slices of 6144 per sample
    const int nl = 32 * C96 * 14 * 56;

    k_init  <<<1, 256, 0, stream>>>(stats);
    k_pack_w<<<(4 * 9216 + 255) / 256, 256, 0, stream>>>(w1, w21, w22, w3, wb);

    k_conv1 <<<ntiles, 192, 0, stream>>>(x, wb, b1, A);               // h = conv1(x)
    k_reduce<<<rg, 256, 0, stream>>>(A, stats + 0);
    k_gngelu<<<eb, 256, 0, stream>>>(A, stats + 0, n1w, n1b, A);      // gelu(gn1(h))
    k_dw3x3 <<<eb, 256, 0, stream>>>(A, dww, dwb, B);                 // dwconv3x3
    k_reduce<<<rg, 256, 0, stream>>>(B, stats + 64);
    k_gngelu<<<eb, 256, 0, stream>>>(B, stats + 64, n2w, n2b, B);     // gelu(gn2(h))
    k_lif   <<<(nl + 255) / 256, 256, 0, stream>>>(B, tau1, vth1, A, 0);  // x_lr
    k_lif   <<<(nl + 255) / 256, 256, 0, stream>>>(B, tau2, vth2, Cc, 1); // x_td
    k_conv2 <<<ntiles, 192, 0, stream>>>(A, Cc, wb + 9216, wb + 18432,
                                         b21, b22, B);                // z = g(..)+g(..)
    k_reduce<<<rg, 256, 0, stream>>>(B, stats + 128);
    k_conv3 <<<ntiles, 192, 0, stream>>>(B, stats + 128, n3w, n3b,
                                         wb + 27648, b3, out);        // conv3(gn3(z))
}
